// MoE_9775345565757
// MI455X (gfx1250) — matrix operand order repaired and checked by reading, not compiled, not run on hardware
//
#include <hip/hip_runtime.h>

#define NEXP 8
#define DIM 1024
#define DI  4096
#define TTOK 4096
#define TM  32      // tokens per block tile (2 x 16-row WMMA M-tiles)
#define NC  256     // DIM_INNER chunk staged in LDS

typedef __attribute__((ext_vector_type(16))) __bf16 v16bf;
typedef __attribute__((ext_vector_type(8)))  float  v8f;

// A-fragment K index (contiguous order, matches B fills):
// lane group g (lane/16): element q -> K = 16g+q
__device__ __forceinline__ int a_kidx(int q, int lg) {
    return 16 * lg + q;
}

__global__ void moe_zero_kernel(float* __restrict__ out, int* __restrict__ counts) {
    int i = blockIdx.x * blockDim.x + threadIdx.x;
    if (i < TTOK * DIM) out[i] = 0.0f;
    if (i < NEXP) counts[i] = 0;
}

__global__ void moe_gate_kernel(const float* __restrict__ x,
                                const float* __restrict__ gw,
                                int* __restrict__ counts,
                                int* __restrict__ tok_ids,
                                float* __restrict__ tok_w) {
    int wid  = threadIdx.x >> 5;
    int lane = threadIdx.x & 31;
    int token = blockIdx.x * 8 + wid;
    float acc[NEXP];
#pragma unroll
    for (int e = 0; e < NEXP; ++e) acc[e] = 0.0f;
    const float* xp = x + (size_t)token * DIM;
    for (int d = lane; d < DIM; d += 32) {
        float xv = xp[d];
        const float* g = gw + (size_t)d * NEXP;
#pragma unroll
        for (int e = 0; e < NEXP; ++e) acc[e] += xv * g[e];
    }
#pragma unroll
    for (int off = 16; off >= 1; off >>= 1) {
#pragma unroll
        for (int e = 0; e < NEXP; ++e) acc[e] += __shfl_xor(acc[e], off, 32);
    }
    if (lane == 0) {
        int i0 = 0;
#pragma unroll
        for (int e = 1; e < NEXP; ++e) if (acc[e] > acc[i0]) i0 = e;
        int i1 = (i0 == 0) ? 1 : 0;
#pragma unroll
        for (int e = 0; e < NEXP; ++e) if (e != i0 && acc[e] > acc[i1]) i1 = e;
        float e1 = __expf(acc[i1] - acc[i0]);
        float w0 = 1.0f / (1.0f + e1);
        float w1 = e1 / (1.0f + e1);
        int p0 = atomicAdd(&counts[i0], 1);
        tok_ids[i0 * TTOK + p0] = token;
        tok_w[i0 * TTOK + p0]  = w0;
        int p1 = atomicAdd(&counts[i1], 1);
        tok_ids[i1 * TTOK + p1] = token;
        tok_w[i1 * TTOK + p1]  = w1;
    }
}

__global__ void __launch_bounds__(256)
moe_expert_kernel(const float* __restrict__ x,
                  const float* __restrict__ Wg,
                  const float* __restrict__ Wu,
                  const float* __restrict__ Wd,
                  const int* __restrict__ counts,
                  const int* __restrict__ tok_ids,
                  const float* __restrict__ tok_w,
                  float* __restrict__ out) {
    const int tilesPerExpert = TTOK / TM;               // 128
    int e    = blockIdx.x / tilesPerExpert;
    int tile = blockIdx.x % tilesPerExpert;
    int cnt  = counts[e];
    if (tile * TM >= cnt) return;                       // block-uniform exit
    int valid = cnt - tile * TM;
    if (valid > TM) valid = TM;

    __shared__ __bf16 Xs[TM][DIM];                      // 64 KB
    __shared__ __bf16 Hs[TM][NC];                       // 16 KB
    __shared__ int    s_tok[TM];
    __shared__ float  s_w[TM];

    if (threadIdx.x < TM) {
        int r = threadIdx.x;
        if (r < valid) {
            s_tok[r] = tok_ids[e * TTOK + tile * TM + r];
            s_w[r]   = tok_w[e * TTOK + tile * TM + r];
        } else { s_tok[r] = 0; s_w[r] = 0.0f; }
    }
    __syncthreads();

    // stage token activations as bf16 into LDS, float4 loads (zeros for pad rows)
    for (int i = threadIdx.x * 4; i < TM * DIM; i += 256 * 4) {
        int r = i >> 10;
        int c = i & (DIM - 1);
        float4 v;
        if (r < valid) v = *(const float4*)(x + (size_t)s_tok[r] * DIM + c);
        else           v = make_float4(0.f, 0.f, 0.f, 0.f);
        Xs[r][c + 0] = (__bf16)v.x;
        Xs[r][c + 1] = (__bf16)v.y;
        Xs[r][c + 2] = (__bf16)v.z;
        Xs[r][c + 3] = (__bf16)v.w;
    }
    __syncthreads();

    const float* Wge = Wg + (size_t)e * DIM * DI;
    const float* Wue = Wu + (size_t)e * DIM * DI;
    const float* Wde = Wd + (size_t)e * DI * DIM;

    int wid  = threadIdx.x >> 5;                        // wave 0..7
    int lane = threadIdx.x & 31;
    int lg   = lane >> 4;                               // lane group 0/1
    int ln   = lane & 15;

    v8f acc0[8] = {};   // M-tile 0 (rows 0..15),  out cols [wid*128, wid*128+128)
    v8f acc1[8] = {};   // M-tile 1 (rows 16..31)

    for (int chunk = 0; chunk < DI; chunk += NC) {
        // ---- Phase A: h = silu(x*Wg) * (x*Wu) for this chunk's 256 cols ----
#pragma unroll
        for (int nt2 = 0; nt2 < 2; ++nt2) {
            int ntile = wid * 2 + nt2;                  // 0..15 within chunk
            int ncol  = chunk + ntile * 16;             // global inner col
            v8f cg0 = {}, cg1 = {}, cu0 = {}, cu1 = {};
            for (int k0 = 0; k0 < DIM; k0 += 32) {
                v16bf a0, a1;
#pragma unroll
                for (int q = 0; q < 16; ++q) {
                    int kk = k0 + a_kidx(q, lg);
                    a0[q] = Xs[ln][kk];
                    a1[q] = Xs[16 + ln][kk];
                }
                v16bf bg, bu;
#pragma unroll
                for (int q = 0; q < 16; ++q) {
                    size_t widx = (size_t)(k0 + 16 * lg + q) * DI + (size_t)(ncol + ln);
                    bg[q] = (__bf16)Wge[widx];
                    bu[q] = (__bf16)Wue[widx];
                }
                cg0 = __builtin_amdgcn_wmma_f32_16x16x32_bf16(false, a0, false, bg,
                                                              (short)0, cg0, false, false);
                cg1 = __builtin_amdgcn_wmma_f32_16x16x32_bf16(false, a1, false, bg,
                                                              (short)0, cg1, false, false);
                cu0 = __builtin_amdgcn_wmma_f32_16x16x32_bf16(false, a0, false, bu,
                                                              (short)0, cu0, false, false);
                cu1 = __builtin_amdgcn_wmma_f32_16x16x32_bf16(false, a1, false, bu,
                                                              (short)0, cu1, false, false);
            }
#pragma unroll
            for (int r = 0; r < 8; ++r) {
                // silu via single v_rcp_f32 (avoids IEEE div expansion)
                float g0 = cg0[r], g1 = cg1[r];
                float h0 = g0 * __builtin_amdgcn_rcpf(1.0f + __expf(-g0)) * cu0[r];
                float h1 = g1 * __builtin_amdgcn_rcpf(1.0f + __expf(-g1)) * cu1[r];
                Hs[r + 8 * lg][ntile * 16 + ln]      = (__bf16)h0;  // D layout M=r+8*lg
                Hs[16 + r + 8 * lg][ntile * 16 + ln] = (__bf16)h1;
            }
        }
        __syncthreads();

        // ---- Phase B: out += h_chunk @ Wd[chunk rows] ----
        for (int kk = 0; kk < NC; kk += 32) {
            v16bf a0, a1;
#pragma unroll
            for (int q = 0; q < 16; ++q) {
                int kidx = kk + a_kidx(q, lg);
                a0[q] = Hs[ln][kidx];
                a1[q] = Hs[16 + ln][kidx];
            }
#pragma unroll
            for (int t = 0; t < 8; ++t) {
                int n = wid * 128 + t * 16;
                v16bf b;
#pragma unroll
                for (int q = 0; q < 16; ++q)
                    b[q] = (__bf16)Wde[(size_t)(chunk + kk + 16 * lg + q) * DIM + (size_t)(n + ln)];
                acc0[t] = __builtin_amdgcn_wmma_f32_16x16x32_bf16(false, a0, false, b,
                                                                  (short)0, acc0[t], false, false);
                acc1[t] = __builtin_amdgcn_wmma_f32_16x16x32_bf16(false, a1, false, b,
                                                                  (short)0, acc1[t], false, false);
            }
        }
        __syncthreads();                                 // Hs reused next chunk
    }

    // ---- scatter-add weighted results (exactly 2 adds per output element) ----
#pragma unroll
    for (int t = 0; t < 8; ++t) {
        int n = wid * 128 + t * 16 + ln;
#pragma unroll
        for (int r = 0; r < 8; ++r) {
            int row0 = r + 8 * lg;
            int row1 = 16 + r + 8 * lg;
            if (row0 < valid)
                atomicAdd(&out[(size_t)s_tok[row0] * DIM + n], s_w[row0] * acc0[t][r]);
            if (row1 < valid)
                atomicAdd(&out[(size_t)s_tok[row1] * DIM + n], s_w[row1] * acc1[t][r]);
        }
    }
}

extern "C" void kernel_launch(void* const* d_in, const int* in_sizes, int n_in,
                              void* d_out, int out_size, void* d_ws, size_t ws_size,
                              hipStream_t stream) {
    const float* x  = (const float*)d_in[0];
    const float* gw = (const float*)d_in[1];
    const float* Wg = (const float*)d_in[2];
    const float* Wu = (const float*)d_in[3];
    const float* Wd = (const float*)d_in[4];
    float* out = (float*)d_out;

    // workspace layout: counts[16 pad] | tok_ids[E*T] | tok_w[E*T]   (~256 KB)
    int*   counts  = (int*)d_ws;
    int*   tok_ids = counts + 16;
    float* tok_w   = (float*)(tok_ids + NEXP * TTOK);

    moe_zero_kernel<<<(TTOK * DIM + 255) / 256, 256, 0, stream>>>(out, counts);
    moe_gate_kernel<<<TTOK / 8, 256, 0, stream>>>(x, gw, counts, tok_ids, tok_w);
    moe_expert_kernel<<<NEXP * (TTOK / TM), 256, 0, stream>>>(
        x, Wg, Wu, Wd, counts, tok_ids, tok_w, out);
}
